// HolographicModel_14259291422732
// MI455X (gfx1250) — matrix operand order repaired and checked by reading, not compiled, not run on hardware
//
#include <hip/hip_runtime.h>
#include <hip/hip_bf16.h>
#include <math.h>

typedef __bf16 bf16_t;
typedef __attribute__((ext_vector_type(16))) __bf16 bf16x16;
typedef __attribute__((ext_vector_type(8)))  __bf16 bf16x8;
typedef __attribute__((ext_vector_type(4)))  __bf16 bf16x4;
typedef __attribute__((ext_vector_type(8)))  float  f32x8;
typedef __attribute__((ext_vector_type(4)))  float  f32x4;

#define D_MODEL 2048
#define M_TOK   4096      // B*T = 2*2048
#define N_FUNC  24        // N_LAYERS*2
#define RANK    16
#define VOCAB   32000
#define EPS_RMS 1.1920929e-07f

static __device__ __forceinline__ bf16_t to_bf16(float f) { return (bf16_t)f; }

// 32-bit LDS byte offset of a generic pointer into static shared memory.
static __device__ __forceinline__ unsigned lds_addr(const void* p) {
    return (unsigned)(unsigned long long)(__attribute__((address_space(3))) const void*)p;
}

// CDNA5 async DMA: global -> LDS, 16B per lane, tracked by ASYNCcnt.
static __device__ __forceinline__ void async_copy_b128(unsigned lds_off, const void* gaddr) {
    asm volatile("global_load_async_to_lds_b128 %0, %1, off"
                 :: "v"(lds_off), "v"(gaddr) : "memory");
}
static __device__ __forceinline__ void wait_async0() {
    asm volatile("s_wait_asynccnt 0x0" ::: "memory");
}

// ---------------------------------------------------------------------------
// W[f, i, j] = (hr[i,j]*pr - hi[i,j]*pi) * scale[f]
// Stored bf16, layout [f][i(out)][j(in)], j contiguous -> GEMM B operand [N,K].
// ---------------------------------------------------------------------------
__global__ __launch_bounds__(256)
void reconstruct_w_kernel(const float* __restrict__ hr,
                          const float* __restrict__ hi,
                          const float* __restrict__ kar,
                          const float* __restrict__ kai,
                          const float* __restrict__ kbr,
                          const float* __restrict__ kbi,
                          const float* __restrict__ scale,
                          bf16_t* __restrict__ W) {
    const int j = blockIdx.x * 16 + (threadIdx.x & 15);
    const int i = blockIdx.y * 16 + (threadIdx.x >> 4);
    const int f = blockIdx.z;

    const float h_r = hr[(size_t)i * D_MODEL + j];
    const float h_i = hi[(size_t)i * D_MODEL + j];

    const float* ar = kar + (size_t)f * RANK * D_MODEL;
    const float* ai = kai + (size_t)f * RANK * D_MODEL;
    const float* br = kbr + (size_t)f * RANK * D_MODEL;
    const float* bi = kbi + (size_t)f * RANK * D_MODEL;

    float pr = 0.f, pi = 0.f;
#pragma unroll
    for (int r = 0; r < RANK; ++r) {
        const float a_r = ar[r * D_MODEL + i];
        const float a_i = ai[r * D_MODEL + i];
        const float b_r = br[r * D_MODEL + j];
        const float b_i = bi[r * D_MODEL + j];
        pr += a_r * b_r + a_i * b_i;
        pi += a_i * b_r - a_r * b_i;
    }
    const float w = (h_r * pr - h_i * pi) * scale[f];
    W[((size_t)f * D_MODEL + i) * D_MODEL + j] = to_bf16(w);
}

// ---------------------------------------------------------------------------
__global__ __launch_bounds__(256)
void embed_kernel(const int* __restrict__ tokens,
                  const float* __restrict__ embed,
                  float* __restrict__ x) {
    const int row = blockIdx.x;
    const int tok = tokens[row];
    const float* src = embed + (size_t)tok * D_MODEL;
    float* dst = x + (size_t)row * D_MODEL;
    for (int c = threadIdx.x; c < D_MODEL; c += 256) dst[c] = src[c];
}

// ---------------------------------------------------------------------------
__global__ __launch_bounds__(256)
void rmsnorm_kernel(const float* __restrict__ x,
                    const float* __restrict__ w,
                    bf16_t* __restrict__ h) {
    __shared__ float red[256];
    const int row = blockIdx.x;
    const float* xr = x + (size_t)row * D_MODEL;

    float v[8];
    float s = 0.f;
#pragma unroll
    for (int i = 0; i < 8; ++i) {
        v[i] = xr[threadIdx.x + i * 256];
        s += v[i] * v[i];
    }
    red[threadIdx.x] = s;
    __syncthreads();
#pragma unroll
    for (int off = 128; off > 0; off >>= 1) {
        if (threadIdx.x < off) red[threadIdx.x] += red[threadIdx.x + off];
        __syncthreads();
    }
    const float rs = rsqrtf(red[0] * (1.0f / D_MODEL) + EPS_RMS);
    bf16_t* hrow = h + (size_t)row * D_MODEL;
#pragma unroll
    for (int i = 0; i < 8; ++i) {
        const int c = threadIdx.x + i * 256;
        hrow[c] = to_bf16(v[i] * rs * w[c]);
    }
}

// ---------------------------------------------------------------------------
// GEMM: Out[M,N] (+)= A[M,K](bf16) x B[N,K]^T via v_wmma_f32_16x16x32_bf16.
// 128x128x32 block tile, 4 waves x (64x64), double-buffered LDS.
// A & B tiles DMA'd with global_load_async_to_lds_b128 (ASYNCcnt).
// B fragments read with plain ds_load_b128 row reads (lane = N column, elems = K).
// EPI: 0 = store, 1 = residual add, 2 = residual += silu(acc)
// BF32: B source is f32 (lm_head_w); converted to bf16 in synchronous staging.
// ---------------------------------------------------------------------------
#define BM 128
#define BN 128
#define BK 32
#define LDT 40   // bf16 elems per LDS tile row (80B, 16B aligned)

template <int EPI, bool BF32>
__global__ __launch_bounds__(128)
void gemm_kernel(const bf16_t* __restrict__ A,
                 const void* __restrict__ Bp,
                 float* __restrict__ Out,
                 int M, int N, int K) {
    __shared__ bf16_t As[2][BM][LDT];
    __shared__ bf16_t Bs[2][BN][LDT];   // natural [n][k] layout

    const int tid  = threadIdx.x;
    const int wave = tid >> 5;
    const int lane = tid & 31;
    const int m0 = blockIdx.y * BM;
    const int n0 = blockIdx.x * BN;
    const int wm = (wave >> 1) * 64;
    const int wn = (wave & 1) * 64;

    const bf16_t* B16 = (const bf16_t*)Bp;
    const float*  B32 = (const float*)Bp;

    f32x8 acc[4][4];
#pragma unroll
    for (int a = 0; a < 4; ++a)
#pragma unroll
        for (int b = 0; b < 4; ++b)
#pragma unroll
            for (int e = 0; e < 8; ++e) acc[a][b][e] = 0.0f;

    auto stage = [&](int buf, int k0) {
        const int chunk = tid & 3;       // 4 x 8 bf16 chunks per row
        const int rbase = tid >> 2;      // 32 rows per pass
#pragma unroll
        for (int p = 0; p < 4; ++p) {
            const int row = rbase + p * 32;
            async_copy_b128(lds_addr(&As[buf][row][chunk * 8]),
                            A + (size_t)(m0 + row) * K + k0 + chunk * 8);
        }
        if (!BF32) {
#pragma unroll
            for (int p = 0; p < 4; ++p) {
                const int row = rbase + p * 32;
                async_copy_b128(lds_addr(&Bs[buf][row][chunk * 8]),
                                B16 + (size_t)(n0 + row) * K + k0 + chunk * 8);
            }
        } else {
            const int kc = tid & 7;      // 4 f32 per chunk
            const int nb = tid >> 3;     // 16 n per pass
#pragma unroll
            for (int p = 0; p < 8; ++p) {
                const int n = nb + p * 16;
                const f32x4 vv = *(const f32x4*)(B32 + (size_t)(n0 + n) * K + k0 + kc * 4);
                bf16x4 cv;
#pragma unroll
                for (int e = 0; e < 4; ++e) cv[e] = to_bf16(vv[e]);
                *(bf16x4*)(&Bs[buf][n][kc * 4]) = cv;
            }
        }
    };

    const int nsteps = K / BK;
    stage(0, 0);

    for (int s = 0; s < nsteps; ++s) {
        wait_async0();
        __syncthreads();
        if (s + 1 < nsteps) stage((s + 1) & 1, (s + 1) * BK);
        const int cur = s & 1;

        // ---- B fragments: direct 16B row reads (lane = N, elems = K) from LDS.
        // Two 16B loads (K 0-15, K 16-31) per 32x16 fragment; wait folded into
        // the same asm block so the WMMA consumer cannot overtake it.
        unsigned ad[8];
#pragma unroll
        for (int ni = 0; ni < 4; ++ni) {
            const int n  = wn + ni * 16 + (lane & 15);
            const int kc = (lane >> 4) * 8;
            ad[2 * ni]     = lds_addr(&Bs[cur][n][kc]);
            ad[2 * ni + 1] = lds_addr(&Bs[cur][n][16 + kc]);
        }
        bf16x8 t0, t1, t2, t3, t4, t5, t6, t7;
        asm volatile(
            "ds_load_b128 %0, %8\n\t"
            "ds_load_b128 %1, %9\n\t"
            "ds_load_b128 %2, %10\n\t"
            "ds_load_b128 %3, %11\n\t"
            "ds_load_b128 %4, %12\n\t"
            "ds_load_b128 %5, %13\n\t"
            "ds_load_b128 %6, %14\n\t"
            "ds_load_b128 %7, %15\n\t"
            "s_wait_dscnt 0x0"
            : "=v"(t0), "=v"(t1), "=v"(t2), "=v"(t3),
              "=v"(t4), "=v"(t5), "=v"(t6), "=v"(t7)
            : "v"(ad[0]), "v"(ad[1]), "v"(ad[2]), "v"(ad[3]),
              "v"(ad[4]), "v"(ad[5]), "v"(ad[6]), "v"(ad[7])
            : "memory");

        bf16x16 bfrag[4];
#pragma unroll
        for (int e = 0; e < 8; ++e) {
            bfrag[0][e] = t0[e]; bfrag[0][e + 8] = t1[e];
            bfrag[1][e] = t2[e]; bfrag[1][e + 8] = t3[e];
            bfrag[2][e] = t4[e]; bfrag[2][e + 8] = t5[e];
            bfrag[3][e] = t6[e]; bfrag[3][e + 8] = t7[e];
        }

        // ---- A fragments (documented 16-bit A swizzle) + MMA
#pragma unroll
        for (int mi = 0; mi < 4; ++mi) {
            const int m   = wm + mi * 16 + (lane & 15);
            const int klo = (lane < 16) ? 0 : 8;
            const bf16x8 lo = *(const bf16x8*)(&As[cur][m][klo]);
            const bf16x8 hi = *(const bf16x8*)(&As[cur][m][klo + 16]);
            bf16x16 afrag;
#pragma unroll
            for (int e = 0; e < 8; ++e) { afrag[e] = lo[e]; afrag[e + 8] = hi[e]; }
#pragma unroll
            for (int ni = 0; ni < 4; ++ni) {
                acc[mi][ni] = __builtin_amdgcn_wmma_f32_16x16x32_bf16(
                    false, afrag, false, bfrag[ni], (short)0, acc[mi][ni], false, false);
            }
        }
    }

    // ---- epilogue (C layout: VGPR v -> M=v (+8 upper half-lanes), lane%16 -> N)
#pragma unroll
    for (int mi = 0; mi < 4; ++mi) {
        const int rbase = m0 + wm + mi * 16 + ((lane >> 4) << 3);
#pragma unroll
        for (int ni = 0; ni < 4; ++ni) {
            const int col = n0 + wn + ni * 16 + (lane & 15);
#pragma unroll
            for (int v = 0; v < 8; ++v) {
                const size_t idx = (size_t)(rbase + v) * N + col;
                const float s = acc[mi][ni][v];
                if (EPI == 0)      Out[idx] = s;
                else if (EPI == 1) Out[idx] += s;
                else               Out[idx] += s / (1.0f + __expf(-s));   // silu
            }
        }
    }
}

// ---------------------------------------------------------------------------
extern "C" void kernel_launch(void* const* d_in, const int* in_sizes, int n_in,
                              void* d_out, int out_size, void* d_ws, size_t ws_size,
                              hipStream_t stream) {
    (void)in_sizes; (void)n_in; (void)out_size; (void)ws_size;

    const int*   tokens     = (const int*)d_in[0];
    const float* hr         = (const float*)d_in[1];
    const float* hi         = (const float*)d_in[2];
    const float* kar        = (const float*)d_in[3];
    const float* kai        = (const float*)d_in[4];
    const float* kbr        = (const float*)d_in[5];
    const float* kbi        = (const float*)d_in[6];
    const float* scale      = (const float*)d_in[7];
    const float* norm_w     = (const float*)d_in[8];
    const float* embed      = (const float*)d_in[9];
    const float* lm_head    = (const float*)d_in[10];
    const float* final_norm = (const float*)d_in[11];

    // workspace layout
    bf16_t* W = (bf16_t*)d_ws;                                   // 24*2048*2048 bf16 = 201.3 MB
    const size_t w_bytes = (size_t)N_FUNC * D_MODEL * D_MODEL * sizeof(bf16_t);
    float* x = (float*)((char*)d_ws + w_bytes);                  // 4096*2048 f32 = 33.6 MB
    const size_t x_bytes = (size_t)M_TOK * D_MODEL * sizeof(float);
    bf16_t* h = (bf16_t*)((char*)x + x_bytes);                   // 4096*2048 bf16 = 16.8 MB

    // 1. materialize all 24 weights in bf16
    reconstruct_w_kernel<<<dim3(D_MODEL / 16, D_MODEL / 16, N_FUNC), 256, 0, stream>>>(
        hr, hi, kar, kai, kbr, kbi, scale, W);

    // 2. embedding gather
    embed_kernel<<<M_TOK, 256, 0, stream>>>(tokens, embed, x);

    // 3. 12 layers: (rmsnorm -> gemm+residual) x2 with silu on the second
    const dim3 gl(D_MODEL / BN, M_TOK / BM);
    for (int l = 0; l < 12; ++l) {
        rmsnorm_kernel<<<M_TOK, 256, 0, stream>>>(x, norm_w + (size_t)(2 * l) * D_MODEL, h);
        gemm_kernel<1, false><<<gl, 128, 0, stream>>>(
            h, W + (size_t)(2 * l) * D_MODEL * D_MODEL, x, M_TOK, D_MODEL, D_MODEL);
        rmsnorm_kernel<<<M_TOK, 256, 0, stream>>>(x, norm_w + (size_t)(2 * l + 1) * D_MODEL, h);
        gemm_kernel<2, false><<<gl, 128, 0, stream>>>(
            h, W + (size_t)(2 * l + 1) * D_MODEL * D_MODEL, x, M_TOK, D_MODEL, D_MODEL);
    }

    // 4. final norm + LM head (B operand = f32 lm_head_w, converted in staging)
    rmsnorm_kernel<<<M_TOK, 256, 0, stream>>>(x, final_norm, h);
    gemm_kernel<0, true><<<dim3(VOCAB / BN, M_TOK / BM), 128, 0, stream>>>(
        h, lm_head, (float*)d_out, M_TOK, VOCAB, D_MODEL);
}
